// GlobalMixture_89472758710312
// MI455X (gfx1250) — matrix operand order repaired and checked by reading, not compiled, not run on hardware
//
#include <hip/hip_runtime.h>

#define EPSF 1e-5f

static constexpr int BB = 32;   // batches
static constexpr int LL = 512;  // tokens (reduction dim K)
static constexpr int DD = 768;  // features (M == N)

typedef float v8f __attribute__((ext_vector_type(8)));
typedef __bf16 v16bf __attribute__((ext_vector_type(16)));
typedef unsigned int v8u __attribute__((ext_vector_type(8)));

static constexpr int BM = 128;
static constexpr int BN = 256;
static constexpr int BK = 32;      // K per chunk (32 tokens)
static constexpr int TR = BK / 2;  // 16 packed k-pair rows per chunk
static constexpr int PM = 132;     // row-major LDS pitch (fallback kernel)
static constexpr int PT = 20;      // column-major LDS pitch (plane kernel)
static constexpr int NTG = LL / BK;  // 16 K-chunks

// ---------------------------------------------------------------------------
// helpers
// ---------------------------------------------------------------------------
__device__ __forceinline__ unsigned pack_hi16(unsigned ua, unsigned ub) {
  return __builtin_amdgcn_perm(ub, ua, 0x07060302u);  // (ub[31:16]<<16)|ua[31:16]
}
__device__ __forceinline__ void split2(float a, float b, unsigned& dh, unsigned& dl) {
  const unsigned ua = __float_as_uint(a), ub = __float_as_uint(b);
  dh = pack_hi16(ua, ub);
  const float ea = a - __uint_as_float(ua & 0xFFFF0000u);
  const float eb = b - __uint_as_float(ub & 0xFFFF0000u);
  dl = pack_hi16(__float_as_uint(ea), __float_as_uint(eb));
}
// async copy, saddr form: one SGPR base + per-thread 32-bit byte offset
__device__ __forceinline__ void async_copy_b128_s(const unsigned* sbase, unsigned voff,
                                                  unsigned lds_byte_addr) {
  asm volatile("global_load_async_to_lds_b128 %0, %1, %2"
               :
               : "v"(lds_byte_addr), "v"(voff), "s"(sbase)
               : "memory");
}
__device__ __forceinline__ void async_wait0() {
  asm volatile("s_wait_asynccnt 0x0" ::: "memory");
}
struct U8q { uint4 lo, hi; };
__device__ __forceinline__ v16bf fragA(const unsigned* plane, int col, int half) {
  U8q s;
  s.lo = *(const uint4*)(plane + col * PT + 4 * half);      // t = 4h..4h+3
  s.hi = *(const uint4*)(plane + col * PT + 8 + 4 * half);  // t = 8+4h..11+4h
  return __builtin_bit_cast(v16bf, s);
}
__device__ __forceinline__ v16bf fragB(const unsigned* plane, int col, int half) {
  U8q s;
  s.lo = *(const uint4*)(plane + col * PT + 4 * half);      // t = 4h..4h+3
  s.hi = *(const uint4*)(plane + col * PT + 8 + 4 * half);  // t = 8+4h..11+4h
  return __builtin_bit_cast(v16bf, s);
}

// ---------------------------------------------------------------------------
// Kernel 1: w[b,l] = sqrt(EPS + sum_d x^2), sw = sqrt(w)
// ---------------------------------------------------------------------------
__global__ __launch_bounds__(256) void token_norm_kernel(const float* __restrict__ x,
                                                         float* __restrict__ w,
                                                         float* __restrict__ sw) {
  const int tok  = (blockIdx.x * blockDim.x + threadIdx.x) >> 5;
  const int lane = threadIdx.x & 31;
  if (tok >= BB * LL) return;
  const float4* xt = (const float4*)(x + (size_t)tok * DD);
  float s = 0.f;
#pragma unroll
  for (int j = 0; j < 6; ++j) {
    float4 v = xt[lane + 32 * j];
    s += v.x * v.x + v.y * v.y + v.z * v.z + v.w * v.w;
  }
#pragma unroll
  for (int off = 16; off > 0; off >>= 1) s += __shfl_xor(s, off, 32);
  if (lane == 0) {
    const float wv = sqrtf(EPSF + s);
    w[tok]  = wv;
    sw[tok] = sqrtf(wv);  // symmetric: out = (diag(sw)X)^T (diag(sw)X)
  }
}

// ---------------------------------------------------------------------------
// Kernel 2 (plane path): Y = diag(sw) X -> k-pair-packed bf16 hi/lo planes in
// chunk-blocked, tq-innermost layout:
//   plane[((b*16+tg)*768 + d)*16 + tq] = bf16 y[tg*32+2tq][d] | bf16 y[..+1]<<16
// One thread per (b,tg,d): 32 coalesced b32 reads, 8 contiguous uint4 writes.
// ---------------------------------------------------------------------------
__global__ __launch_bounds__(256) void make_planes_kernel(const float* __restrict__ x,
                                                          const float* __restrict__ sw,
                                                          unsigned* __restrict__ Phi,
                                                          unsigned* __restrict__ Plo) {
  const int bx = blockIdx.x;              // 1536 blocks = 3 * 16 * 32
  const int d  = (bx % 3) * 256 + threadIdx.x;
  const int tg = (bx / 3) & 15;
  const int b  = bx / 48;

  const float*  xc   = x + (size_t)(b * LL + tg * BK) * DD + d;
  const float*  swc  = sw + b * LL + tg * BK;
  const size_t  obase = ((size_t)(b * NTG + tg) * DD + d) * 16;

#pragma unroll
  for (int q = 0; q < 4; ++q) {
    float y[8];
#pragma unroll
    for (int j = 0; j < 8; ++j)
      y[j] = swc[8 * q + j] * xc[(size_t)(8 * q + j) * DD];
    uint4 dh, dl;
    split2(y[0], y[1], dh.x, dl.x);
    split2(y[2], y[3], dh.y, dl.y);
    split2(y[4], y[5], dh.z, dl.z);
    split2(y[6], y[7], dh.w, dl.w);
    *(uint4*)&Phi[obase + 4 * q] = dh;
    *(uint4*)&Plo[obase + 4 * q] = dl;
  }
}

// ---------------------------------------------------------------------------
// Kernel 3 (plane path): out[b] = Y^T Y via V_WMMA_F32_16X16X32_BF16, bf16x3.
// LDS is column-major [col][tq] (pitch 20): each fragment = 2 ds_load_b128.
// Fill = pure async b128 copy (saddr form). Two-pass to stay under 256 VGPRs.
// ---------------------------------------------------------------------------
__global__ __launch_bounds__(256) void gram_planes_kernel(const unsigned* __restrict__ Phi,
                                                          const unsigned* __restrict__ Plo,
                                                          float* __restrict__ out) {
  __shared__ unsigned Acol[2][BM * PT];  // 2 planes x 128 cols x 16(+4) tq
  __shared__ unsigned Bcol[2][BN * PT];  // 2 planes x 256 cols

  const int b  = blockIdx.y;
  const int MT = DD / BM;
  const int m0 = (blockIdx.x % MT) * BM;
  const int n0 = (blockIdx.x / MT) * BN;

  const int tid  = threadIdx.x;
  const int lane = tid & 31;
  const int wid  = tid >> 5;
  const int wm   = wid & 1;
  const int wn   = wid >> 1;
  const int half = lane >> 4;
  const int idx  = lane & 15;

  v8f acc[4][4];
#pragma unroll
  for (int mi = 0; mi < 4; ++mi)
#pragma unroll
    for (int ni = 0; ni < 4; ++ni) acc[mi][ni] = (v8f)0.0f;

  for (int tg = 0; tg < NTG; ++tg) {
    const unsigned* gH = Phi + (size_t)(b * NTG + tg) * DD * 16;
    const unsigned* gL = Plo + (size_t)(b * NTG + tg) * DD * 16;
    __syncthreads();
    // ---- A tile: 128 cols x 4 quads x 2 planes ---------------------------
#pragma unroll
    for (int it = 0; it < 2; ++it) {
      const int s   = tid + it * 256;  // 0..511
      const int col = s >> 2;
      const int q   = s & 3;
      const unsigned voff = (unsigned)(((m0 + col) * 16 + 4 * q) * 4);  // bytes
      const unsigned dst  = (unsigned)(col * PT + 4 * q) * 4;
      async_copy_b128_s(gH, voff, (unsigned)(size_t)Acol[0] + dst);
      async_copy_b128_s(gL, voff, (unsigned)(size_t)Acol[1] + dst);
    }
    // ---- B tile: 256 cols x 4 quads x 2 planes ---------------------------
#pragma unroll
    for (int it = 0; it < 4; ++it) {
      const int s   = tid + it * 256;  // 0..1023
      const int col = s >> 2;
      const int q   = s & 3;
      const unsigned voff = (unsigned)(((n0 + col) * 16 + 4 * q) * 4);
      const unsigned dst  = (unsigned)(col * PT + 4 * q) * 4;
      async_copy_b128_s(gH, voff, (unsigned)(size_t)Bcol[0] + dst);
      async_copy_b128_s(gL, voff, (unsigned)(size_t)Bcol[1] + dst);
    }
    async_wait0();
    __syncthreads();

    // ---- pass 1: Ahi x (Bhi, Blo) --------------------------------------
    {
      v16bf ah[4];
#pragma unroll
      for (int mi = 0; mi < 4; ++mi)
        ah[mi] = fragA(Acol[0], wm * 64 + mi * 16 + idx, half);
#pragma unroll
      for (int ni = 0; ni < 4; ++ni) {
        const int col = wn * 64 + ni * 16 + idx;
        const v16bf bh = fragB(Bcol[0], col, half);
        const v16bf bl = fragB(Bcol[1], col, half);
#pragma unroll
        for (int mi = 0; mi < 4; ++mi) {
          acc[mi][ni] = __builtin_amdgcn_wmma_f32_16x16x32_bf16(
              false, ah[mi], false, bh, (short)0, acc[mi][ni], false, false);
          acc[mi][ni] = __builtin_amdgcn_wmma_f32_16x16x32_bf16(
              false, ah[mi], false, bl, (short)0, acc[mi][ni], false, false);
        }
      }
    }
    // ---- pass 2: Alo x Bhi ----------------------------------------------
    {
      v16bf al[4];
#pragma unroll
      for (int mi = 0; mi < 4; ++mi)
        al[mi] = fragA(Acol[1], wm * 64 + mi * 16 + idx, half);
#pragma unroll
      for (int ni = 0; ni < 4; ++ni) {
        const v16bf bh = fragB(Bcol[0], wn * 64 + ni * 16 + idx, half);
#pragma unroll
        for (int mi = 0; mi < 4; ++mi)
          acc[mi][ni] = __builtin_amdgcn_wmma_f32_16x16x32_bf16(
              false, al[mi], false, bh, (short)0, acc[mi][ni], false, false);
      }
    }
  }

#pragma unroll
  for (int mi = 0; mi < 4; ++mi) {
#pragma unroll
    for (int ni = 0; ni < 4; ++ni) {
      const int n = n0 + wn * 64 + ni * 16 + idx;
#pragma unroll
      for (int r = 0; r < 8; ++r) {
        const int m = m0 + wm * 64 + mi * 16 + r + 8 * half;
        out[((size_t)b * DD + m) * DD + n] = acc[mi][ni][r];
      }
    }
  }
}

// ---------------------------------------------------------------------------
// Fallback (small ws_size): in-kernel conversion bf16x3 GEMM
// ---------------------------------------------------------------------------
__global__ __launch_bounds__(256) void gram_inline_kernel(const float* __restrict__ x,
                                                          const float* __restrict__ w,
                                                          float* __restrict__ out) {
  __shared__ unsigned AH[TR * PM], AL[TR * PM];
  __shared__ unsigned BH[2][TR * PM], BL[2][TR * PM];

  const int b  = blockIdx.y;
  const int MT = DD / BM;
  const int m0 = (blockIdx.x % MT) * BM;
  const int n0 = (blockIdx.x / MT) * BN;

  const int tid  = threadIdx.x;
  const int lane = tid & 31;
  const int wid  = tid >> 5;
  const int wm   = wid & 1;
  const int wn   = wid >> 1;
  const int hb   = wn >> 1;
  const int nb   = (wn & 1) * 64;
  const int half = lane >> 4;
  const int idx  = lane & 15;

  const float* xb = x + (size_t)b * LL * DD;
  const float* wb = w + b * LL;

  v8f acc[4][4];
#pragma unroll
  for (int mi = 0; mi < 4; ++mi)
#pragma unroll
    for (int ni = 0; ni < 4; ++ni) acc[mi][ni] = (v8f)0.0f;

  for (int l0 = 0; l0 < LL; l0 += BK) {
    __syncthreads();
#pragma unroll
    for (int it = 0; it < 2; ++it) {
      const int s  = tid + it * 256;
      const int t  = s >> 5;
      const int m4 = (s & 31) << 2;
      const float* g0 = xb + (size_t)(l0 + 2 * t) * DD + m0 + m4;
      float4 a0 = *(const float4*)g0;
      float4 a1 = *(const float4*)(g0 + DD);
      const float2 wv = *(const float2*)&wb[l0 + 2 * t];
      a0.x *= wv.x; a0.y *= wv.x; a0.z *= wv.x; a0.w *= wv.x;
      a1.x *= wv.y; a1.y *= wv.y; a1.z *= wv.y; a1.w *= wv.y;
      uint4 dh, dl;
      split2(a0.x, a1.x, dh.x, dl.x);
      split2(a0.y, a1.y, dh.y, dl.y);
      split2(a0.z, a1.z, dh.z, dl.z);
      split2(a0.w, a1.w, dh.w, dl.w);
      *(uint4*)&AH[t * PM + m4] = dh;
      *(uint4*)&AL[t * PM + m4] = dl;
    }
#pragma unroll
    for (int it = 0; it < 4; ++it) {
      const int s   = tid + it * 256;
      const int h   = s >> 9;
      const int rem = s & 511;
      const int t   = rem >> 5;
      const int n4  = (rem & 31) << 2;
      const float* g0 = xb + (size_t)(l0 + 2 * t) * DD + n0 + h * 128 + n4;
      const float4 b0 = *(const float4*)g0;
      const float4 b1 = *(const float4*)(g0 + DD);
      uint4 dh, dl;
      split2(b0.x, b1.x, dh.x, dl.x);
      split2(b0.y, b1.y, dh.y, dl.y);
      split2(b0.z, b1.z, dh.z, dl.z);
      split2(b0.w, b1.w, dh.w, dl.w);
      *(uint4*)&BH[h][t * PM + n4] = dh;
      *(uint4*)&BL[h][t * PM + n4] = dl;
    }
    __syncthreads();

    v8u ahu[4], alu[4];
#pragma unroll
    for (int mi = 0; mi < 4; ++mi) {
      const int col = wm * 64 + mi * 16 + idx;
#pragma unroll
      for (int j = 0; j < 8; ++j) {
        const int t = (j & 3) + half * 4 + ((j >> 2) * 8);
        ahu[mi][j] = AH[t * PM + col];
        alu[mi][j] = AL[t * PM + col];
      }
    }
#pragma unroll
    for (int ni = 0; ni < 4; ++ni) {
      const int col = nb + ni * 16 + idx;
      v8u bhu, blu;
#pragma unroll
      for (int j = 0; j < 8; ++j) {
        const int t = (j & 3) + half * 4 + ((j >> 2) * 8);
        bhu[j] = BH[hb][t * PM + col];
        blu[j] = BL[hb][t * PM + col];
      }
      const v16bf bh = __builtin_bit_cast(v16bf, bhu);
      const v16bf bl = __builtin_bit_cast(v16bf, blu);
#pragma unroll
      for (int mi = 0; mi < 4; ++mi) {
        const v16bf ah = __builtin_bit_cast(v16bf, ahu[mi]);
        const v16bf al = __builtin_bit_cast(v16bf, alu[mi]);
        acc[mi][ni] = __builtin_amdgcn_wmma_f32_16x16x32_bf16(
            false, ah, false, bh, (short)0, acc[mi][ni], false, false);
        acc[mi][ni] = __builtin_amdgcn_wmma_f32_16x16x32_bf16(
            false, al, false, bh, (short)0, acc[mi][ni], false, false);
        acc[mi][ni] = __builtin_amdgcn_wmma_f32_16x16x32_bf16(
            false, ah, false, bl, (short)0, acc[mi][ni], false, false);
      }
    }
  }

#pragma unroll
  for (int mi = 0; mi < 4; ++mi) {
#pragma unroll
    for (int ni = 0; ni < 4; ++ni) {
      const int n = n0 + wn * 64 + ni * 16 + idx;
#pragma unroll
      for (int r = 0; r < 8; ++r) {
        const int m = m0 + wm * 64 + mi * 16 + r + 8 * half;
        out[((size_t)b * DD + m) * DD + n] = acc[mi][ni][r];
      }
    }
  }
}

// ---------------------------------------------------------------------------
extern "C" void kernel_launch(void* const* d_in, const int* in_sizes, int n_in,
                              void* d_out, int out_size, void* d_ws, size_t ws_size,
                              hipStream_t stream) {
  const float* x = (const float*)d_in[0];  // [32, 512, 768] fp32
  float* out     = (float*)d_out;          // [32, 768, 768] fp32

  float* wtok  = (float*)d_ws;                          // 64 KB
  float* swtok = (float*)((char*)d_ws + (64 << 10));    // 64 KB
  const size_t PLANE_DW   = (size_t)BB * (LL / 2) * DD; // 6.29M dwords per plane
  const size_t NEED_BYTES = (size_t)(128 << 10) + 2 * PLANE_DW * 4;

  token_norm_kernel<<<dim3((BB * LL) / 8), dim3(256), 0, stream>>>(x, wtok, swtok);

  dim3 grid((DD / BM) * (DD / BN), BB);
  if (ws_size >= NEED_BYTES) {
    unsigned* Phi = (unsigned*)((char*)d_ws + (128 << 10));
    unsigned* Plo = Phi + PLANE_DW;
    make_planes_kernel<<<dim3(3 * NTG * BB), dim3(256), 0, stream>>>(x, swtok, Phi, Plo);
    gram_planes_kernel<<<grid, dim3(256), 0, stream>>>(Phi, Plo, out);
  } else {
    gram_inline_kernel<<<grid, dim3(256), 0, stream>>>(x, wtok, out);
  }
}
